// GRU_63625645523590
// MI455X (gfx1250) — matrix operand order repaired and checked by reading, not compiled, not run on hardware
//
#include <hip/hip_runtime.h>
#include <hip/hip_bf16.h>
#include <cstdint>

// ---------------- problem constants ----------------
#define TT 512
#define BB 64
#define II 512
#define HH 1024
#define OO 512
#define K1 (HH + II)   // 1536 : [h | x] depth
#define N1 (2 * HH)    // 2048 : [z | r] width
#define RPW 8          // batch rows per recurrent workgroup
#define NWG_REC (BB / RPW)  // 8 persistent workgroups
#define RECTHREADS 1024     // 32 waves
#define NT1 4               // GEMM1 N-tiles per wave (32 waves * 4 * 16 = 2048)
#define NT2 2               // GEMM2 N-tiles per wave (32 waves * 2 * 16 = 1024)

typedef __attribute__((ext_vector_type(16))) __bf16 bf16x16;
typedef __attribute__((ext_vector_type(8)))  __bf16 bf16x8;
typedef __attribute__((ext_vector_type(8)))  float  v8f;

// ---------------- helpers ----------------
__device__ __forceinline__ __bf16 f2bf(float f) {
  unsigned u = __builtin_bit_cast(unsigned, f);
  u = (u + 0x7FFFu + ((u >> 16) & 1u)) >> 16;
  unsigned short s = (unsigned short)u;
  return __builtin_bit_cast(__bf16, s);
}
__device__ __forceinline__ float sigmoidf_(float x) {
  return 1.0f / (1.0f + __expf(-x));
}
__device__ __forceinline__ bf16x16 cat8(bf16x8 lo, bf16x8 hi) {
  return __builtin_shufflevector(lo, hi, 0,1,2,3,4,5,6,7,8,9,10,11,12,13,14,15);
}
__device__ __forceinline__ bf16x8 zero_bf8() {
  uint4 z = {0u, 0u, 0u, 0u};
  return __builtin_bit_cast(bf16x8, z);
}
__device__ __forceinline__ v8f wmma_bf16(bf16x16 a, bf16x16 b, v8f c) {
  return __builtin_amdgcn_wmma_f32_16x16x32_bf16(
      /*neg_a=*/false, a, /*neg_b=*/false, b,
      /*c_mod=*/(short)0, c, /*reuse_a=*/false, /*reuse_b=*/false);
}

// ---------------- workspace layout (bytes) ----------------
// Xbf   : T*B*I bf16  = 33,554,432
// Wg1   : [K1][N1]    =  6,291,456   rows0..1023=[Wzh|Wrh], rows1024..1535=[Wzx|Wrx]
// Wg2   : [K1][HH]    =  3,145,728   rows0..1023=Whh, rows1024..1535=Whx
// WoutT : [HH][OO]    =  1,048,576   WoutT[k][o] = W_out[o][k]
// Hall  : T*B*H bf16  = 67,108,864
#define OFF_XBF   ((size_t)0)
#define OFF_WG1   ((size_t)33554432)
#define OFF_WG2   ((size_t)39845888)
#define OFF_WOUTT ((size_t)42991616)
#define OFF_HALL  ((size_t)44040192)

// ================= kernel 1: fp32 -> bf16 input conversion =================
__global__ void conv_x_kernel(const float* __restrict__ X, __bf16* __restrict__ Xbf, int n) {
  for (int i = blockIdx.x * blockDim.x + threadIdx.x; i < n; i += gridDim.x * blockDim.x)
    Xbf[i] = f2bf(X[i]);
}

// ================= kernel 2: pack + convert weights =================
__global__ void conv_w_kernel(const float* __restrict__ Wzh, const float* __restrict__ Wzx,
                              const float* __restrict__ Wrh, const float* __restrict__ Wrx,
                              const float* __restrict__ Whh, const float* __restrict__ Whx,
                              const float* __restrict__ Wout,
                              __bf16* __restrict__ Wg1, __bf16* __restrict__ Wg2,
                              __bf16* __restrict__ WoutT) {
  const int NG1 = K1 * N1;        // 3,145,728
  const int NG2 = K1 * HH;        // 1,572,864
  const int NGO = HH * OO;        //   524,288
  const int total = NG1 + NG2 + NGO;
  for (int i = blockIdx.x * blockDim.x + threadIdx.x; i < total; i += gridDim.x * blockDim.x) {
    if (i < NG1) {
      int r = i / N1, c = i % N1;
      float v;
      if (r < HH) v = (c < HH) ? Wzh[r * HH + c] : Wrh[r * HH + (c - HH)];
      else {
        int rx = r - HH;
        v = (c < HH) ? Wzx[rx * HH + c] : Wrx[rx * HH + (c - HH)];
      }
      Wg1[i] = f2bf(v);
    } else if (i < NG1 + NG2) {
      int j = i - NG1;
      int r = j / HH, c = j % HH;
      float v = (r < HH) ? Whh[r * HH + c] : Whx[(r - HH) * HH + c];
      Wg2[j] = f2bf(v);
    } else {
      int j = i - NG1 - NG2;
      int k = j / OO, o = j % OO;
      WoutT[j] = f2bf(Wout[o * HH + k]);
    }
  }
}

// ================= kernel 3: persistent GRU recurrence =================
// grid = 8 workgroups x 1024 threads (32 waves, 8/SIMD). Each WG owns 8 batch
// rows and runs the full 512-step loop; batch rows are independent => no grid
// sync, only workgroup barriers.
// LDS: A [16][K1] bf16 (48KB, [h|x] operand), Hs [16][HH] f32 (64KB), Zs (64KB)
#define SMEM_A   0
#define SMEM_HS  49152
#define SMEM_ZS  (49152 + 65536)
#define SMEM_REC (49152 + 65536 + 65536)  // 180,224 B (<320KB/WG on CDNA5)

__global__ void __launch_bounds__(RECTHREADS, 1)
gru_rec_kernel(const __bf16* __restrict__ Xbf, const float* __restrict__ state,
               const float* __restrict__ b_z, const float* __restrict__ b_r,
               const float* __restrict__ b_h,
               const __bf16* __restrict__ Wg1, const __bf16* __restrict__ Wg2,
               __bf16* __restrict__ Hall, float* __restrict__ hfin) {
  extern __shared__ char smem[];
  __bf16* A  = (__bf16*)(smem + SMEM_A);
  float*  Hs = (float*)(smem + SMEM_HS);
  float*  Zs = (float*)(smem + SMEM_ZS);

  const int tid  = threadIdx.x;
  const int wave = tid >> 5;
  const int lane = tid & 31;
  const int rlo  = lane & 15;   // A-fragment row, C-fragment column
  const int rhi  = lane >> 4;   // selects K sub-chunk / row-half
  const int b0   = blockIdx.x * RPW;

  // ---- cache per-lane biases (constant over all 512 steps) ----
  float bias1[NT1];   // z-cols for waves 0..15, r-cols for waves 16..31
  float bias2[NT2];
#pragma unroll
  for (int nt = 0; nt < NT1; ++nt) {
    int col = (wave * NT1 + nt) * 16 + rlo;
    bias1[nt] = (col < HH) ? b_z[col] : b_r[col - HH];
  }
#pragma unroll
  for (int nt = 0; nt < NT2; ++nt) {
    int col = (wave * NT2 + nt) * 16 + rlo;
    bias2[nt] = b_h[col];
  }

  // ---- init: h state (pad rows 8..15 with zeros) ----
  for (int idx = tid; idx < 16 * HH; idx += RECTHREADS) {
    int r = idx >> 10, c = idx & (HH - 1);
    float hv = (r < RPW) ? state[(b0 + r) * HH + c] : 0.0f;
    Hs[idx] = hv;
    A[r * K1 + c] = f2bf(hv);
  }
  // ---- init: x_0 into A[:, HH:HH+II]; pad rows zeroed once, never rewritten
  for (int idx = tid; idx < 16 * (II / 8); idx += RECTHREADS) {
    int r = idx >> 6, c8 = idx & 63;
    bf16x8 v = zero_bf8();
    if (r < RPW) v = *(const bf16x8*)(Xbf + ((size_t)(b0 + r)) * II + c8 * 8);
    *(bf16x8*)(A + r * K1 + HH + c8 * 8) = v;
  }
  __syncthreads();

  for (int t = 0; t < TT; ++t) {
    // ======== GEMM1: [h|x](16xK1) @ Wg1(K1xN1) -> z|r pre-acts; 4 N-tiles/wave
    v8f acc[NT1] = {};
    for (int kt = 0; kt < K1 / 32; ++kt) {
      const int k0 = kt * 32;
      bf16x8 alo = *(const bf16x8*)(A + rlo * K1 + k0 + rhi * 8);
      bf16x8 ahi = *(const bf16x8*)(A + rlo * K1 + k0 + rhi * 8 + 16);
      bf16x16 a = cat8(alo, ahi);
      const __bf16* bp = Wg1 + (size_t)k0 * N1 + rlo;
      __builtin_prefetch(bp + (size_t)32 * N1, 0, 3);  // next K-tile row (near)
#pragma unroll
      for (int nt = 0; nt < NT1; ++nt) {
        bf16x16 b;
#pragma unroll
        for (int s = 0; s < 16; ++s)
          b[s] = bp[(size_t)(rhi * 8 + (s & 7) + (s >> 3) * 16) * N1 + (wave * NT1 + nt) * 16];
        acc[nt] = wmma_bf16(a, b, acc[nt]);
      }
    }
    __syncthreads();

    // ======== activations: z -> LDS(f32); r -> hr = r*h -> A[:,0:HH] (bf16)
#pragma unroll
    for (int nt = 0; nt < NT1; ++nt) {
      const int col = (wave * NT1 + nt) * 16 + rlo;
#pragma unroll
      for (int i = 0; i < 8; ++i) {
        const int row = i + 8 * rhi;
        float v = acc[nt][i] + bias1[nt];
        if (col < HH) {
          Zs[row * HH + col] = sigmoidf_(v);
        } else {
          const int c = col - HH;
          float rr = sigmoidf_(v);
          A[row * K1 + c] = f2bf(rr * Hs[row * HH + c]);
        }
      }
    }
    __syncthreads();

    // ======== GEMM2: [h*r|x](16xK1) @ Wg2(K1xHH) -> candidate; 2 N-tiles/wave
    v8f acc2[NT2] = {};
    for (int kt = 0; kt < K1 / 32; ++kt) {
      const int k0 = kt * 32;
      bf16x8 alo = *(const bf16x8*)(A + rlo * K1 + k0 + rhi * 8);
      bf16x8 ahi = *(const bf16x8*)(A + rlo * K1 + k0 + rhi * 8 + 16);
      bf16x16 a = cat8(alo, ahi);
      const __bf16* bp = Wg2 + (size_t)k0 * HH + rlo;
      __builtin_prefetch(bp + (size_t)32 * HH, 0, 3);
#pragma unroll
      for (int nt = 0; nt < NT2; ++nt) {
        bf16x16 b;
#pragma unroll
        for (int s = 0; s < 16; ++s)
          b[s] = bp[(size_t)(rhi * 8 + (s & 7) + (s >> 3) * 16) * HH + (wave * NT2 + nt) * 16];
        acc2[nt] = wmma_bf16(a, b, acc2[nt]);
      }
    }
    __syncthreads();

    // ======== update: h = (1-z)h + z*tanh(.); write LDS + Hall
#pragma unroll
    for (int nt = 0; nt < NT2; ++nt) {
      const int col = (wave * NT2 + nt) * 16 + rlo;
#pragma unroll
      for (int i = 0; i < 8; ++i) {
        const int row = i + 8 * rhi;
        float ht = tanhf(acc2[nt][i] + bias2[nt]);
        float z  = Zs[row * HH + col];
        float h0 = Hs[row * HH + col];
        float hn = (1.0f - z) * h0 + z * ht;
        Hs[row * HH + col] = hn;
        __bf16 hb = f2bf(hn);
        A[row * K1 + col] = hb;
        if (row < RPW)
          Hall[((size_t)t * BB + b0 + row) * HH + col] = hb;
      }
    }
    // ======== stage x_{t+1} into A[:, HH:] via async global->LDS DMA ========
    // 512 chunks of 16B; waves 0..15 issue one GLOBAL_LOAD_ASYNC_TO_LDS_B128
    // per lane (ASYNCcnt-tracked), then fence with s_wait_asynccnt before the
    // barrier. Pad rows (8..15) keep their init-time zeros.
    if (t + 1 < TT) {
      if (tid < RPW * (II / 8)) {   // wave-uniform: waves 0..15
        const int r  = tid >> 6;
        const int c8 = tid & 63;
        const __bf16* gsrc =
            Xbf + ((size_t)(t + 1) * BB + b0 + r) * II + c8 * 8;
        const unsigned lds_off =
            (unsigned)(SMEM_A + (r * K1 + HH + c8 * 8) * 2);
        asm volatile("global_load_async_to_lds_b128 %0, %1, off"
                     :
                     : "v"(lds_off), "v"((unsigned long long)(uintptr_t)gsrc)
                     : "memory");
      }
      asm volatile("s_wait_asynccnt 0" ::: "memory");
    }
    __syncthreads();
  }

  // ---- h_final ----
  for (int idx = tid; idx < RPW * HH; idx += RECTHREADS) {
    int r = idx >> 10, c = idx & (HH - 1);
    hfin[(size_t)(b0 + r) * HH + c] = Hs[r * HH + c];
  }
}

// ================= kernel 4: Y = Hall(32768x1024) @ WoutT(1024x512) + b_out
// grid = 2048 WGs x 256 thr (8 waves). Each wave: one 16-row x 64-col strip.
__global__ void __launch_bounds__(256)
out_gemm_kernel(const __bf16* __restrict__ Hall, const __bf16* __restrict__ WoutT,
                const float* __restrict__ b_out, float* __restrict__ Y) {
  const int wave = threadIdx.x >> 5;
  const int lane = threadIdx.x & 31;
  const int rlo = lane & 15, rhi = lane >> 4;
  const int rt = blockIdx.x;       // row tile: 16 rows of T*B
  const int n0 = wave * 64;        // 8 waves cover 512 cols

  v8f acc[4] = {};
  const __bf16* arow = Hall + ((size_t)rt * 16 + rlo) * HH;
  for (int kt = 0; kt < HH / 32; ++kt) {
    const int k0 = kt * 32;
    bf16x8 alo = *(const bf16x8*)(arow + k0 + rhi * 8);
    bf16x8 ahi = *(const bf16x8*)(arow + k0 + rhi * 8 + 16);
    bf16x16 a = cat8(alo, ahi);
    const __bf16* bp = WoutT + (size_t)k0 * OO + rlo;
    __builtin_prefetch(bp + (size_t)32 * OO, 0, 3);
#pragma unroll
    for (int nt = 0; nt < 4; ++nt) {
      bf16x16 b;
#pragma unroll
      for (int s = 0; s < 16; ++s)
        b[s] = bp[(size_t)(rhi * 8 + (s & 7) + (s >> 3) * 16) * OO + n0 + nt * 16];
      acc[nt] = wmma_bf16(a, b, acc[nt]);
    }
  }
#pragma unroll
  for (int nt = 0; nt < 4; ++nt) {
    const int col = n0 + nt * 16 + rlo;
    const float bb = b_out[col];
#pragma unroll
    for (int i = 0; i < 8; ++i) {
      const int row = rt * 16 + i + 8 * rhi;
      Y[(size_t)row * OO + col] = acc[nt][i] + bb;
    }
  }
}

// ================= host launcher =================
extern "C" void kernel_launch(void* const* d_in, const int* in_sizes, int n_in,
                              void* d_out, int out_size, void* d_ws, size_t ws_size,
                              hipStream_t stream) {
  (void)in_sizes; (void)n_in; (void)out_size; (void)ws_size;
  const float* X     = (const float*)d_in[0];
  const float* state = (const float*)d_in[1];
  const float* Wzh   = (const float*)d_in[2];
  const float* Wzx   = (const float*)d_in[3];
  const float* bz    = (const float*)d_in[4];
  const float* Wrh   = (const float*)d_in[5];
  const float* Wrx   = (const float*)d_in[6];
  const float* br    = (const float*)d_in[7];
  const float* Whh   = (const float*)d_in[8];
  const float* Whx   = (const float*)d_in[9];
  const float* bh    = (const float*)d_in[10];
  const float* Wout  = (const float*)d_in[11];
  const float* bout  = (const float*)d_in[12];

  char* ws = (char*)d_ws;
  __bf16* Xbf   = (__bf16*)(ws + OFF_XBF);
  __bf16* Wg1   = (__bf16*)(ws + OFF_WG1);
  __bf16* Wg2   = (__bf16*)(ws + OFF_WG2);
  __bf16* WoutT = (__bf16*)(ws + OFF_WOUTT);
  __bf16* Hall  = (__bf16*)(ws + OFF_HALL);

  float* Y    = (float*)d_out;
  float* hfin = Y + (size_t)TT * BB * OO;

  conv_x_kernel<<<2048, 256, 0, stream>>>(X, Xbf, TT * BB * II);
  conv_w_kernel<<<2048, 256, 0, stream>>>(Wzh, Wzx, Wrh, Wrx, Whh, Whx, Wout,
                                          Wg1, Wg2, WoutT);
  gru_rec_kernel<<<NWG_REC, RECTHREADS, SMEM_REC, stream>>>(Xbf, state, bz, br, bh,
                                                            Wg1, Wg2, Hall, hfin);
  out_gemm_kernel<<<(TT * BB) / 16, 256, 0, stream>>>(Hall, WoutT, bout, Y);
}
